// E74Layer_7335804141913
// MI455X (gfx1250) — matrix operand order repaired and checked by reading, not compiled, not run on hardware
//
#include <hip/hip_runtime.h>
#include <math.h>

// ---------- problem constants (match reference) ----------
#define T_LEN   2048
#define BATCH   8
#define DIM_    1024
#define DINNER  2048
#define NSTATE  64
#define ROWS    (T_LEN * BATCH)      // 16384 token rows
#define NPROJ   (4 * NSTATE)         // 256 = k|v|q|z stacked

typedef __attribute__((ext_vector_type(16))) __bf16          v16bf;
typedef __attribute__((ext_vector_type(16))) unsigned short  v16u;
typedef __attribute__((ext_vector_type(8)))  unsigned short  v8u;
typedef __attribute__((ext_vector_type(8)))  float           v8f;

union Frag { v16u u; v16bf b; };

__device__ inline unsigned short f2bf(float f) {
  // round-to-nearest-even f32 -> bf16
  unsigned int u = __float_as_uint(f);
  u += 0x7FFFu + ((u >> 16) & 1u);
  return (unsigned short)(u >> 16);
}

// ---------------------------------------------------------------------------
// WMMA GEMM, bf16 operands already in memory:
//   C[M,N] = A[M,K] * B^T   with A [M,K] bf16 row-major, B [N,K] bf16 row-major
// One wave per 16x64 C panel: 1 A-fragment feeds 4 B-fragments / 4 WMMAs per
// 32-deep K step. Fragment lane layouts (CDNA5 ISA 7.12.2, wave32) make each
// fragment exactly two 16-byte loads per lane:
//   A: M = lane&15 ; elems 0..7 -> K = kk + 8*hi + [0..7] ; 8..15 -> +16
//   B: N = lane&15 ; elems 0..7 -> K = kk + 8*hi + [0..7] ; 8..15 -> +16
//   C: elem r -> M = r + 8*hi, N = lane&15
// ---------------------------------------------------------------------------
template <bool OUTBF>
__global__ __launch_bounds__(32)
void wmma_gemm_bt(const unsigned short* __restrict__ A,
                  const unsigned short* __restrict__ B,
                  void* __restrict__ Cv, int M, int N, int K) {
  const int ntn  = N >> 6;                       // 64-wide N panels
  const int m0   = (blockIdx.x / ntn) << 4;
  const int n0   = (blockIdx.x % ntn) << 6;
  const int lane = threadIdx.x & 31;
  const int lo   = lane & 15;
  const int hi   = lane >> 4;

  const unsigned short* __restrict__ ar = A + (size_t)(m0 + lo) * K;
  const unsigned short* __restrict__ br = B + (size_t)(n0 + lo) * K;

  v8f acc[4] = {{}, {}, {}, {}};
  for (int kk = 0; kk < K; kk += 32) {
    const v8u a0 = *(const v8u*)(ar + kk + (hi << 3));
    const v8u a1 = *(const v8u*)(ar + kk + 16 + (hi << 3));
    Frag fa;
    fa.u = __builtin_shufflevector(a0, a1, 0, 1, 2, 3, 4, 5, 6, 7,
                                   8, 9, 10, 11, 12, 13, 14, 15);
#pragma unroll
    for (int j = 0; j < 4; ++j) {
      const unsigned short* bj = br + (size_t)(j << 4) * K + kk + (hi << 3);
      const v8u b0 = *(const v8u*)(bj);
      const v8u b1 = *(const v8u*)(bj + 16);
      Frag fb;
      fb.u = __builtin_shufflevector(b0, b1, 0, 1, 2, 3, 4, 5, 6, 7,
                                     8, 9, 10, 11, 12, 13, 14, 15);
      acc[j] = __builtin_amdgcn_wmma_f32_16x16x32_bf16(
          false, fa.b, false, fb.b, (short)0, acc[j], false, false);
    }
  }
#pragma unroll
  for (int j = 0; j < 4; ++j) {
#pragma unroll
    for (int r = 0; r < 8; ++r) {
      const size_t idx = (size_t)(m0 + r + (hi << 3)) * N + n0 + (j << 4) + lo;
      if (OUTBF) ((unsigned short*)Cv)[idx] = f2bf(acc[j][r]);
      else       ((float*)Cv)[idx]          = acc[j][r];
    }
  }
}

// ---------------------------------------------------------------------------
// One-time bf16 conversion passes (bandwidth-bound, ~100 MB total @ 23.3 TB/s)
// ---------------------------------------------------------------------------
__global__ void cvt_bf16(const float* __restrict__ in,
                         unsigned short* __restrict__ out, int n) {
  int i = blockIdx.x * blockDim.x + threadIdx.x;
  if (i < n) out[i] = f2bf(in[i]);
}

// W_in [R=DINNER, C=DIM] -> W_inT bf16 [DIM, DINNER]  (out[c*R + r] = in[r*C + c])
__global__ void cvt_t_bf16(const float* __restrict__ in,
                           unsigned short* __restrict__ out, int R, int C) {
  int i = blockIdx.x * blockDim.x + threadIdx.x;
  if (i >= R * C) return;
  int r = i / C, c = i - r * C;
  out[(size_t)c * R + r] = f2bf(in[i]);
}

// Stack W_k|W_v|W_q|W_z into contiguous bf16 [256, 2048].
__global__ void concat4_bf16(const float* __restrict__ a, const float* __restrict__ b,
                             const float* __restrict__ c, const float* __restrict__ d,
                             unsigned short* __restrict__ o) {
  const int per = NSTATE * DINNER;  // 131072
  int i = blockIdx.x * blockDim.x + threadIdx.x;
  if (i >= 4 * per) return;
  int sel = i / per, j = i - sel * per;
  const float* src = (sel == 0) ? a : (sel == 1) ? b : (sel == 2) ? c : d;
  o[i] = f2bf(src[j]);
}

// ---------------------------------------------------------------------------
// Delta-rule scan (latency-bound critical path). One block per batch, 64
// threads; thread n owns state row S[n][0..63] in VGPRs. kvqz row layout:
// k(64)|v(64)|q(64)|z(64) f32. Double-buffered LDS broadcast of k,q -> one
// barrier per step; next token prefetched into registers. y written as bf16.
// ---------------------------------------------------------------------------
__global__ __launch_bounds__(64)
void delta_scan(const float* __restrict__ kvqz, unsigned short* __restrict__ y) {
  const int b = blockIdx.x;
  const int n = threadIdx.x;

  __shared__ float kbuf[2][NSTATE];
  __shared__ float qbuf[2][NSTATE];

  float S[NSTATE];
#pragma unroll
  for (int m = 0; m < NSTATE; ++m) S[m] = 0.0f;

  const float* p0 = kvqz + (size_t)b * NPROJ;
  float kc = p0[n], vc = p0[NSTATE + n], qc = p0[2 * NSTATE + n],
        zc = p0[3 * NSTATE + n];

  for (int t = 0; t < T_LEN; ++t) {
    const int buf = t & 1;
    kbuf[buf][n] = kc;
    qbuf[buf][n] = qc;
    const float v = vc, z = zc;
    __syncthreads();

    if (t + 1 < T_LEN) {
      const float* pn = kvqz + ((size_t)(t + 1) * BATCH + b) * NPROJ;
      kc = pn[n]; vc = pn[NSTATE + n];
      qc = pn[2 * NSTATE + n]; zc = pn[3 * NSTATE + n];
    }
    if (t + 2 < T_LEN) {  // global_prefetch_b8 warm-ahead
      __builtin_prefetch(kvqz + ((size_t)(t + 2) * BATCH + b) * NPROJ + n, 0, 1);
    }

    float pred = 0.0f, ksum = 0.0f;
#pragma unroll
    for (int m = 0; m < NSTATE; ++m) {
      const float km = kbuf[buf][m];
      pred = fmaf(S[m], km, pred);
      ksum = fmaf(km, km, ksum);
    }
    const float inv  = 1.0f / (sqrtf(ksum) + 1e-6f);
    const float coef = (v - pred * inv) * inv;

    float yacc = 0.0f;
#pragma unroll
    for (int m = 0; m < NSTATE; ++m) {
      const float s = fmaf(coef, kbuf[buf][m], S[m]);
      S[m] = s;
      yacc = fmaf(s, qbuf[buf][m], yacc);
    }

    const float sig = 1.0f / (1.0f + expf(-z));
    y[((size_t)t * BATCH + b) * NSTATE + n] = f2bf(tanhf(yacc) * sig);
  }
}

// ---------------------------------------------------------------------------
extern "C" void kernel_launch(void* const* d_in, const int* in_sizes, int n_in,
                              void* d_out, int out_size, void* d_ws,
                              size_t ws_size, hipStream_t stream) {
  const float* x     = (const float*)d_in[0];  // [T,B,DIM]
  const float* W_in  = (const float*)d_in[1];  // [DINNER, DIM]
  const float* W_k   = (const float*)d_in[2];  // [NSTATE, DINNER]
  const float* W_v   = (const float*)d_in[3];
  const float* W_q   = (const float*)d_in[4];
  const float* W_z   = (const float*)d_in[5];
  const float* W_out = (const float*)d_in[6];  // [DIM, NSTATE]
  float* out = (float*)d_out;                  // [T,B,DIM]

  // ---- workspace carve-up (f32 region first, then bf16 regions; all 16B
  //      aligned since every element count is a multiple of 8). ~58 MB total.
  float* kvqz = (float*)d_ws;                               // [16384,256] f32
  unsigned short* xbf   = (unsigned short*)(kvqz + (size_t)ROWS * NPROJ);
  unsigned short* winT  = xbf   + (size_t)ROWS * DIM_;      // [DIM, DINNER]
  unsigned short* wcat  = winT  + (size_t)DIM_ * DINNER;    // [256, 2048]
  unsigned short* woutb = wcat  + (size_t)NPROJ * DINNER;   // [DIM, 64]
  unsigned short* wc    = woutb + (size_t)DIM_ * NSTATE;    // [256, 1024]
  unsigned short* ybf   = wc    + (size_t)NPROJ * DIM_;     // [16384, 64]

  // 1) one-time bf16 conversions / packing
  {
    int n = ROWS * DIM_;
    cvt_bf16<<<(n + 255) / 256, 256, 0, stream>>>(x, xbf, n);
  }
  cvt_t_bf16<<<(DINNER * DIM_ + 255) / 256, 256, 0, stream>>>(W_in, winT,
                                                              DINNER, DIM_);
  concat4_bf16<<<(4 * NSTATE * DINNER + 255) / 256, 256, 0, stream>>>(
      W_k, W_v, W_q, W_z, wcat);
  {
    int n = DIM_ * NSTATE;
    cvt_bf16<<<(n + 255) / 256, 256, 0, stream>>>(W_out, woutb, n);
  }

  // 2) fold in_proj into head projections:
  //    Wc[256,1024] = Wcat[256,2048] @ W_inT[1024,2048]^T   (bf16 out)
  wmma_gemm_bt<true><<<dim3((NPROJ / 16) * (DIM_ / 64)), 32, 0, stream>>>(
      wcat, winT, wc, NPROJ, DIM_, DINNER);
  // 3) token projections: kvqz[16384,256] = x[16384,1024] @ Wc^T  (f32 out)
  wmma_gemm_bt<false><<<dim3((ROWS / 16) * (NPROJ / 64)), 32, 0, stream>>>(
      xbf, wc, kvqz, ROWS, NPROJ, DIM_);
  // 4) sequential delta-rule scan (critical path), writes y as bf16
  delta_scan<<<BATCH, NSTATE, 0, stream>>>(kvqz, ybf);
  // 5) out_proj: out[16384,1024] = y[16384,64] @ W_out^T  (f32 out)
  wmma_gemm_bt<false><<<dim3((ROWS / 16) * (DIM_ / 64)), 32, 0, stream>>>(
      ybf, woutb, out, ROWS, DIM_, NSTATE);
}
